// QuantLinearLUT_53687091200511
// MI455X (gfx1250) — matrix operand order repaired and checked by reading, not compiled, not run on hardware
//
#include <hip/hip_runtime.h>
#include <hip/hip_bf16.h>
#include <cstdint>
#include <cstddef>

#define IN_DIM  4096
#define OUT_DIM 11008
#define M_ROWS  8192

typedef __attribute__((ext_vector_type(16))) __bf16 v16bf;
typedef __attribute__((ext_vector_type(8)))  float  v8f;
typedef unsigned int u32x4 __attribute__((ext_vector_type(4)));
typedef int          i32x8 __attribute__((ext_vector_type(8)));
typedef int          i32x4 __attribute__((ext_vector_type(4)));

#if defined(__gfx1250__) && __has_builtin(__builtin_amdgcn_tensor_load_to_lds)
#define HAVE_TDM 1
#else
#define HAVE_TDM 0
#endif

__device__ __forceinline__ unsigned short f2bf_rne(float f) {
  unsigned u = __float_as_uint(f);
  u += 0x7FFFu + ((u >> 16) & 1u);   // round-to-nearest-even
  return (unsigned short)(u >> 16);
}
__device__ __forceinline__ float bf2f(unsigned short h) {
  return __uint_as_float(((unsigned)h) << 16);
}

// ---------------------------------------------------------------------------
// Async global->LDS fallback path (ASYNCcnt-tracked).
// ---------------------------------------------------------------------------
__device__ __forceinline__ void async_b128(unsigned lds_addr, const void* base,
                                           unsigned byte_off) {
  asm volatile("global_load_async_to_lds_b128 %0, %1, %2"
               :: "v"(lds_addr), "v"(byte_off), "s"(base) : "memory");
}
__device__ __forceinline__ void wait_async0() {
  asm volatile("s_wait_asynccnt 0x0" ::: "memory");
}

#if HAVE_TDM
// ---------------------------------------------------------------------------
// Tensor Data Mover: one descriptor moves a 2-D tile (tile_d0 x tile_d1,
// 2-byte elements) from a strided tensor into LDS, inserting 4 dwords of
// padding after every 16 dwords (= one 32-elem row) -> 80 B LDS row stride.
// D# layout per CDNA5 ISA ch.8: group0 = flags/lds/global addr (type=2),
// group1 = sizes/strides/pad config. Groups 2/3 (and the trailing group of
// the 6-arg builtin) unused for 2-D tiles.
// ---------------------------------------------------------------------------
__device__ __forceinline__ void tdm_load_2d(unsigned lds_addr, const void* gaddr,
                                            unsigned tensor_d0, unsigned tensor_d1,
                                            unsigned tile_d0, unsigned tile_d1,
                                            unsigned stride0) {
  const unsigned long long ga = (unsigned long long)gaddr;
  u32x4 g0;
  g0[0] = 1u;                                             // count=1 (valid user D#)
  g0[1] = lds_addr;                                       // LDS byte address
  g0[2] = (unsigned)ga;                                   // global_addr[31:0]
  g0[3] = (unsigned)((ga >> 32) & 0x01FFFFFFu) | (2u << 30); // addr[56:32] | type=2
  i32x8 g1;
  // data_size=1 (2B) <<16 | pad_enable<<20 | pad_interval=3 (16dw) <<22
  // | pad_amount=3 (4dw) <<25 ; workgroup_mask=0
  g1[0] = (int)((1u << 16) | (1u << 20) | (3u << 22) | (3u << 25));
  g1[1] = (int)(tensor_d0 << 16);                         // tensor_dim0[15:0] @bit48
  g1[2] = (int)((tensor_d0 >> 16) | (tensor_d1 << 16));   // dim0 hi | dim1 lo
  g1[3] = (int)((tensor_d1 >> 16) | (tile_d0 << 16));     // dim1 hi | tile_dim0
  g1[4] = (int)(tile_d1 & 0xFFFFu);                       // tile_dim1 (tile_dim2=0)
  g1[5] = (int)stride0;                                   // tensor_dim0_stride lo
  g1[6] = 0;                                              // stride0 hi | stride1 lo
  g1[7] = 0;
  const i32x4 z4 = {0, 0, 0, 0};
  const i32x8 z8 = {0, 0, 0, 0, 0, 0, 0, 0};
  __builtin_amdgcn_tensor_load_to_lds(g0, g1, z4, z4, z8, 0);
}
#endif

// ---------------------------------------------------------------------------
// Kernel 1: x (f32) -> x_bf16, 8 elements per thread, b128 store
// ---------------------------------------------------------------------------
__global__ void k_cvt_x(const float* __restrict__ x, unsigned short* __restrict__ xb) {
  const int t = blockIdx.x * blockDim.x + threadIdx.x;
  const size_t base = (size_t)t * 8;
  float4 a = *(const float4*)(x + base);
  float4 b = *(const float4*)(x + base + 4);
  union { unsigned short s[8]; uint4 q; } u;
  u.s[0] = f2bf_rne(a.x); u.s[1] = f2bf_rne(a.y);
  u.s[2] = f2bf_rne(a.z); u.s[3] = f2bf_rne(a.w);
  u.s[4] = f2bf_rne(b.x); u.s[5] = f2bf_rne(b.y);
  u.s[6] = f2bf_rne(b.z); u.s[7] = f2bf_rne(b.w);
  *(uint4*)(xb + base) = u.q;
}

// ---------------------------------------------------------------------------
// Kernel 2: LUT dequant. qweight:[IN/8][OUT] int32 (8 nibbles along IN).
// ---------------------------------------------------------------------------
__global__ void k_dequant(const int* __restrict__ qw,
                          const float* __restrict__ lut,
                          unsigned short* __restrict__ wb) {
  const int idx = blockIdx.x * 256 + threadIdx.x;   // == r*OUT + o
  const int o = idx % OUT_DIM;
  const int r = idx / OUT_DIM;
  const unsigned w = (unsigned)qw[idx];
  const float* l = lut + (size_t)o * 16;
  union { unsigned short s[8]; uint4 q; } u;
#pragma unroll
  for (int j = 0; j < 8; ++j)
    u.s[j] = f2bf_rne(l[(w >> (4 * j)) & 0xF]);
  *(uint4*)(wb + (size_t)o * IN_DIM + r * 8) = u.q;
}

// ---------------------------------------------------------------------------
// Kernel 3: CSR corrections. One thread owns one output row -> serial RMW.
// ---------------------------------------------------------------------------
__global__ void k_csr(const int* __restrict__ rp0, const int* __restrict__ c0,
                      const float* __restrict__ v0,
                      const int* __restrict__ rp1, const int* __restrict__ c1,
                      const float* __restrict__ v1,
                      unsigned short* __restrict__ wb) {
  const int o = blockIdx.x * 256 + threadIdx.x;     // 43*256 == 11008 exactly
  if (o >= OUT_DIM) return;
  unsigned short* row = wb + (size_t)o * IN_DIM;
  for (int i = rp0[o]; i < rp0[o + 1]; ++i) {
    int c = c0[i];
    row[c] = f2bf_rne(bf2f(row[c]) + v0[i]);
  }
  for (int i = rp1[o]; i < rp1[o + 1]; ++i) {
    int c = c1[i];
    row[c] = f2bf_rne(bf2f(row[c]) + v1[i]);
  }
}

// ---------------------------------------------------------------------------
// Kernel 4: GEMM  y[m,o] = sum_k xb[m,k]*wb[o,k] + bias[o]
// Block: 256 threads = 8 waves, tile 128(M) x 128(N), K-step 32.
// Wave tile 64(M) x 32(N): 8 accumulators, 2 B-frags x 4 A-frags per step.
// Double-buffered LDS filled by the Tensor Data Mover (wave 0 issues two
// 2-D descriptors per K-tile, overlapped with WMMAs; TENSORcnt + barrier).
// LDS row stride 40 halfwords (80 B) via TDM pad injection.
// ---------------------------------------------------------------------------
#define MT  128
#define NTB 128
#define KT  32
#define LDSS (KT + 8)     // 40 ushorts = 80 B per row

union Frag { v16bf v; uint4 q[2]; };

__global__ __launch_bounds__(256) void k_gemm(
    const unsigned short* __restrict__ xb,
    const unsigned short* __restrict__ wb,
    const float* __restrict__ bias,
    float* __restrict__ y) {
  __shared__ __align__(16) unsigned short As[2 * MT * LDSS];   // 20.0 KB
  __shared__ __align__(16) unsigned short Bs[2 * NTB * LDSS];  // 20.0 KB

  const int tid  = threadIdx.x;
  const int lane = tid & 31;
  const int wave = tid >> 5;
  const int half = lane >> 4;      // 0..1
  const int l16  = lane & 15;      // 0..15

  const int mBase = blockIdx.y * MT;
  const int nBase = blockIdx.x * NTB;
  const int wm = (wave >> 2) * 64; // wave M offset: 0 or 64
  const int wn = (wave & 3) * 32;  // wave N offset: 0,32,64,96

  const unsigned asBase = (unsigned)(uintptr_t)As;  // LDS byte addr = low 32b
  const unsigned bsBase = (unsigned)(uintptr_t)Bs;

  // stage K-tile `kt` into buffer `buf`
  auto stage = [&](int buf, int kt) {
#if HAVE_TDM
    if (wave == 0) {
      tdm_load_2d(asBase + (unsigned)(buf * MT * LDSS * 2),
                  xb + (size_t)mBase * IN_DIM + kt,
                  IN_DIM, M_ROWS, KT, MT, IN_DIM);
      tdm_load_2d(bsBase + (unsigned)(buf * NTB * LDSS * 2),
                  wb + (size_t)nBase * IN_DIM + kt,
                  IN_DIM, OUT_DIM, KT, NTB, IN_DIM);
    }
#else
#pragma unroll
    for (int u = 0; u < 2; ++u) {
      const int e = tid + u * 256;
      const int row = e >> 2, col = (e & 3) * 8;
      async_b128(asBase + (unsigned)(((buf * MT + row) * LDSS + col) * 2), xb,
                 (unsigned)(((mBase + row) * IN_DIM + kt + col) * 2));
      async_b128(bsBase + (unsigned)(((buf * NTB + row) * LDSS + col) * 2), wb,
                 (unsigned)(((nBase + row) * IN_DIM + kt + col) * 2));
    }
#endif
  };
  auto stage_wait = [&]() {
#if HAVE_TDM
    __builtin_amdgcn_s_wait_tensorcnt(0);  // no-op for non-issuing waves
#else
    wait_async0();
#endif
    __syncthreads();
  };

  // ---- prologue ----
  stage(0, 0);
  stage_wait();

  v8f acc[2][4];
#pragma unroll
  for (int j = 0; j < 2; ++j)
#pragma unroll
    for (int t = 0; t < 4; ++t) acc[j][t] = (v8f){};

  for (int kt = 0, it = 0; kt < IN_DIM; kt += KT, ++it) {
    const int cur = it & 1;
    const int kn = kt + KT;
    if (kn < IN_DIM) stage(cur ^ 1, kn);   // overlaps with WMMAs below

    // ---- compute on current buffer ----
    const unsigned short* Ab = &As[cur * MT * LDSS];
    const unsigned short* Bb = &Bs[cur * NTB * LDSS];

    Frag fb[2], fa[4];
    // B fragments: lane holds column n; k = half*16 + (0..15)
#pragma unroll
    for (int j = 0; j < 2; ++j) {
      const unsigned short* bp = &Bb[(wn + j * 16 + l16) * LDSS + half * 16];
      fb[j].q[0] = *(const uint4*)(bp);
      fb[j].q[1] = *(const uint4*)(bp + 8);
    }
    // A fragments: lane holds row m; k = half*16 + (0..15)
#pragma unroll
    for (int t = 0; t < 4; ++t) {
      const unsigned short* ap = &Ab[(wm + t * 16 + l16) * LDSS + half * 16];
      fa[t].q[0] = *(const uint4*)(ap);
      fa[t].q[1] = *(const uint4*)(ap + 8);
    }
#pragma unroll
    for (int j = 0; j < 2; ++j)
#pragma unroll
      for (int t = 0; t < 4; ++t)
        acc[j][t] = __builtin_amdgcn_wmma_f32_16x16x32_bf16(
            false, fa[t].v, false, fb[j].v, (short)0, acc[j][t], false, false);

    stage_wait();
  }

  // ---- epilogue: D[r,lane] -> m = r + 8*half, n = l16 per 16x16 subtile ----
#pragma unroll
  for (int j = 0; j < 2; ++j) {
    const int gn = nBase + wn + j * 16 + l16;
    const float bv = bias[gn];
#pragma unroll
    for (int t = 0; t < 4; ++t) {
#pragma unroll
      for (int r = 0; r < 8; ++r) {
        const int gm = mBase + wm + t * 16 + r + half * 8;
        y[(size_t)gm * OUT_DIM + gn] = acc[j][t][r] + bv;
      }
    }
  }
}

// ---------------------------------------------------------------------------
// Host launcher. Inputs (setup_inputs order):
// 0:x 1:qweight 2:lookup_table 3:bias 4:outlier_rows 5:outlier_cols
// 6:outlier_vals 7:sensitive_rows 8:sensitive_cols 9:sensitive_vals
// Workspace: [ x_bf16 : 8192*4096 u16 ][ W_bf16 : 11008*4096 u16 ] = 150 MiB
// ---------------------------------------------------------------------------
extern "C" void kernel_launch(void* const* d_in, const int* in_sizes, int n_in,
                              void* d_out, int out_size, void* d_ws, size_t ws_size,
                              hipStream_t stream) {
  const float* x    = (const float*)d_in[0];
  const int*   qw   = (const int*)d_in[1];
  const float* lut  = (const float*)d_in[2];
  const float* bias = (const float*)d_in[3];
  const int*   orow = (const int*)d_in[4];
  const int*   ocol = (const int*)d_in[5];
  const float* oval = (const float*)d_in[6];
  const int*   srow = (const int*)d_in[7];
  const int*   scol = (const int*)d_in[8];
  const float* sval = (const float*)d_in[9];
  float* y = (float*)d_out;

  unsigned short* xb = (unsigned short*)d_ws;
  unsigned short* wb = xb + (size_t)M_ROWS * IN_DIM;

  k_cvt_x<<<16384, 256, 0, stream>>>(x, xb);
  k_dequant<<<22016, 256, 0, stream>>>(qw, lut, wb);
  k_csr<<<43, 256, 0, stream>>>(orow, ocol, oval, srow, scol, sval, wb);
  k_gemm<<<dim3(OUT_DIM / NTB, M_ROWS / MT), 256, 0, stream>>>(xb, wb, bias, y);
}
